// EncoderBlock_32598801777316
// MI455X (gfx1250) — matrix operand order repaired and checked by reading, not compiled, not run on hardware
//
#include <hip/hip_runtime.h>
#include <hip/hip_bf16.h>
#include <math.h>

// ---------------------------------------------------------------------------
// Problem constants
// ---------------------------------------------------------------------------
#define BATCH   16
#define CH      256
#define RES     56
#define WS      7
#define HEADS   8
#define HD      32          // head dim
#define GROUPS  16
#define MLPD    1024
#define NTOK    49          // tokens per window
#define NPAD    64          // padded tokens (4 x 16)
#define NWIN    1024        // BATCH * 8 * 8
#define SPAT    3136        // 56*56
#define PIX     (BATCH*SPAT)   // 50176 rows for MLP GEMMs

typedef __bf16 bf16_t;
typedef __attribute__((ext_vector_type(16))) __bf16 v16bf;
typedef __attribute__((ext_vector_type(8)))  float  v8f;

// ---------------------------------------------------------------------------
// WMMA helpers (CDNA5 v_wmma_f32_16x16x32_bf16, wave32)
// ---------------------------------------------------------------------------
__device__ __forceinline__ v8f wmma_bf16(v16bf a, v16bf b, v8f c) {
    return __builtin_amdgcn_wmma_f32_16x16x32_bf16(
        false, a, false, b, (short)0, c, false, false);
}

// A fragment: 16x32 (MxK) bf16, row-major source with leading dim lda.
// lane&15 = M row; lanes 0-15 hold K 0-15; lanes 16-31 hold K 16-31.
__device__ __forceinline__ v16bf load_a_frag(const bf16_t* __restrict__ A, int lda) {
    const int lane = threadIdx.x & 31;
    const int row  = lane & 15;
    const int kb   = (lane >> 4) << 4;   // 0 or 16
    v16bf a;
#pragma unroll
    for (int i = 0; i < 8; ++i) {
        const int k = kb + 2 * i;
        a[2 * i]     = A[row * lda + k];
        a[2 * i + 1] = A[row * lda + k + 1];
    }
    return a;
}

// B fragment from an [N][K] row-major matrix (weights stored [out][in]).
// lane&15 = N col; lanes 0-15 hold K 0-15, lanes 16-31 hold K 16-31 (paired).
__device__ __forceinline__ v16bf load_b_frag_nk(const bf16_t* __restrict__ W, int ldw) {
    const int lane = threadIdx.x & 31;
    const int n    = lane & 15;
    const int kb   = (lane >> 4) << 4;   // 0 or 16
    v16bf b;
#pragma unroll
    for (int i = 0; i < 8; ++i) {
        b[2 * i]     = W[n * ldw + kb + 2 * i];
        b[2 * i + 1] = W[n * ldw + kb + 2 * i + 1];
    }
    return b;
}

// B fragment from a [K][N] row-major matrix (e.g. V: 64xHD).
__device__ __forceinline__ v16bf load_b_frag_kn(const bf16_t* __restrict__ Bm, int ldb) {
    const int lane = threadIdx.x & 31;
    const int n    = lane & 15;
    const int kb   = (lane >> 4) << 4;
    v16bf b;
#pragma unroll
    for (int i = 0; i < 8; ++i) {
        b[2 * i]     = Bm[(kb + 2 * i) * ldb + n];
        b[2 * i + 1] = Bm[(kb + 2 * i + 1) * ldb + n];
    }
    return b;
}

// C/D layout: VGPR r -> row (r + 8*(lane>=16)), col = lane&15.
__device__ __forceinline__ void store_c_bf16(bf16_t* __restrict__ D, int ldd, v8f c) {
    const int lane = threadIdx.x & 31;
    const int col  = lane & 15;
    const int rb   = (lane >> 4) << 3;
#pragma unroll
    for (int r = 0; r < 8; ++r) D[(rb + r) * ldd + col] = (bf16_t)c[r];
}

// ---------------------------------------------------------------------------
// CDNA5 async global->LDS copy (GLOBAL_LOAD_ASYNC_TO_LDS_B128, ASYNCcnt).
// GVS addressing: mem = SGPR base + 32-bit per-lane offset. VDST carries the
// per-lane LDS byte offset. No VGPR data round trip; wave-level overlap.
// ---------------------------------------------------------------------------
__device__ __forceinline__ unsigned lds_off_u32(void* p) {
    // generic -> LDS(addrspace 3) -> 32-bit group-segment byte offset
    return (unsigned)(uintptr_t)(__attribute__((address_space(3))) void*)p;
}
__device__ __forceinline__ void async_b128(unsigned ldsOff, const void* gbase, unsigned gOff) {
    asm volatile("global_load_async_to_lds_b128 %0, %1, %2 offset:0"
                 :: "v"(ldsOff), "v"(gOff), "s"(gbase) : "memory");
}
__device__ __forceinline__ void wait_async0() {
    asm volatile("s_wait_asynccnt 0" ::: "memory");
}

// Stage a 64x(256 bf16) row-major tile (row pitch rowBytes in global) into LDS
// with padded stride 264 halves. 2048 x 16B chunks across 256 threads.
__device__ __forceinline__ void stage_tile_async(bf16_t* lds, const bf16_t* gsrc,
                                                 unsigned rowBytes) {
    const unsigned base = lds_off_u32(lds);
    for (int idx = threadIdx.x; idx < 2048; idx += 256) {
        const unsigned row = idx >> 5, wv = idx & 31;
        async_b128(base + row * (264u * 2u) + wv * 16u, gsrc, row * rowBytes + wv * 16u);
    }
    wait_async0();
}

// ---------------------------------------------------------------------------
// Workspace layout
// ---------------------------------------------------------------------------
#define ALIGN256(x) ((((size_t)(x)) + 255) & ~(size_t)255)
static const size_t OFF_QKVW   = 0;                                        // 768*256 bf16
static const size_t OFF_PROJW  = ALIGN256(OFF_QKVW  + (size_t)768*256*2);
static const size_t OFF_W1     = ALIGN256(OFF_PROJW + (size_t)256*256*2);
static const size_t OFF_W2     = ALIGN256(OFF_W1    + (size_t)1024*256*2);
static const size_t OFF_BIAS   = ALIGN256(OFF_W2    + (size_t)256*1024*2); // 8*49*49 f32
static const size_t OFF_QSCALE = ALIGN256(OFF_BIAS  + (size_t)8*49*49*4);  // 8 f32
static const size_t OFF_QKVN   = ALIGN256(OFF_QSCALE + 64);                // 1024*3*8*64*32 bf16
static const size_t OFF_ATTN   = ALIGN256(OFF_QKVN  + (size_t)NWIN*3*HEADS*NPAD*HD*2);
//   OFF_ATTN region (32 MB) is reused later as the [PIX][256] bf16 MLP input.
static const size_t OFF_Y0     = ALIGN256(OFF_ATTN  + (size_t)NWIN*NPAD*CH*2);
//   OFF_Y0 (51 MB f32) is reused as t = h + mlp_out for GN2.
static const size_t OFF_H      = ALIGN256(OFF_Y0    + (size_t)BATCH*CH*SPAT*4);
static const size_t OFF_M1     = ALIGN256(OFF_H     + (size_t)BATCH*CH*SPAT*4);

// ---------------------------------------------------------------------------
// Kernel 0: weight conversion + bias gather + logit-scale fold
// ---------------------------------------------------------------------------
__global__ __launch_bounds__(256) void k_prep(
    const float* __restrict__ qkv_w, const float* __restrict__ proj_w,
    const float* __restrict__ w1,    const float* __restrict__ w2,
    const float* __restrict__ bias_table, const int* __restrict__ rel_index,
    const float* __restrict__ logit_scale,
    bf16_t* __restrict__ qkvw, bf16_t* __restrict__ projw,
    bf16_t* __restrict__ w1b,  bf16_t* __restrict__ w2b,
    float* __restrict__ biasw, float* __restrict__ qscale) {
    const int tid = blockIdx.x * blockDim.x + threadIdx.x;
    const int np  = gridDim.x * blockDim.x;
    for (int i = tid; i < 768 * 256;  i += np) qkvw[i]  = (bf16_t)qkv_w[i];
    for (int i = tid; i < 256 * 256;  i += np) projw[i] = (bf16_t)proj_w[i];
    for (int i = tid; i < 1024 * 256; i += np) w1b[i]   = (bf16_t)w1[i];
    for (int i = tid; i < 256 * 1024; i += np) w2b[i]   = (bf16_t)w2[i];
    for (int i = tid; i < HEADS * NTOK * NTOK; i += np) {
        const int h = i / (NTOK * NTOK), nm = i % (NTOK * NTOK);
        biasw[i] = bias_table[rel_index[nm] * HEADS + h];
    }
    if (tid < HEADS) qscale[tid] = __expf(fminf(logit_scale[tid], 4.6051702f));
}

// ---------------------------------------------------------------------------
// Kernel 1: window partition + QKV GEMM + q/k row L2-normalize (scale folded)
// One workgroup per window; wave w handles head w for q, k and v.
// ---------------------------------------------------------------------------
__global__ __launch_bounds__(256) void k_qkv(
    const float* __restrict__ x, const float* __restrict__ qkv_b,
    const bf16_t* __restrict__ qkvw, const float* __restrict__ qscale,
    bf16_t* __restrict__ qkvn) {
    __shared__ bf16_t xw[NPAD * 264];   // padded lda to de-conflict LDS banks
    const int win  = blockIdx.x;
    const int bi   = win >> 6, widx = win & 63;
    const int wy   = widx >> 3, wx = widx & 7;
    const float* xb = x + (size_t)bi * CH * SPAT;

    for (int idx = threadIdx.x; idx < NPAD * CH; idx += 256) {
        const int n = idx >> 8, c = idx & 255;
        float v = 0.f;
        if (n < NTOK) {
            const int ty = n / WS, tx = n % WS;
            v = xb[(size_t)c * SPAT + (wy * WS + ty) * RES + (wx * WS + tx)];
        }
        xw[n * 264 + c] = (bf16_t)v;
    }
    __syncthreads();

    const int head = threadIdx.x >> 5;
    const int lane = threadIdx.x & 31;
    const int col  = lane & 15;
    const int half = lane >> 4;

    for (int mat = 0; mat < 3; ++mat) {                 // q, k, v
        const int e0 = mat * CH + head * HD;            // output row in qkv_w
        const float qs = (mat == 0) ? qscale[head] : 1.f;
        bf16_t* dst = qkvn + (((size_t)win * 3 + mat) * HEADS + head) * (NPAD * HD);
        __builtin_prefetch(qkvw + (size_t)e0 * 256, 0, 1);   // L2-resident weight panel
        for (int mt = 0; mt < 4; ++mt) {                // 4 token tiles of 16
            v8f c0 = {}, c1 = {};
#pragma unroll
            for (int k = 0; k < 8; ++k) {               // K = 256
                const v16bf a  = load_a_frag(&xw[mt * 16 * 264 + k * 32], 264);
                const v16bf b0 = load_b_frag_nk(qkvw + (size_t)e0 * 256 + k * 32, 256);
                const v16bf b1 = load_b_frag_nk(qkvw + (size_t)(e0 + 16) * 256 + k * 32, 256);
                c0 = wmma_bf16(a, b0, c0);
                c1 = wmma_bf16(a, b1, c1);
            }
            const float bb0 = qkv_b[e0 + col];
            const float bb1 = qkv_b[e0 + 16 + col];
#pragma unroll
            for (int r = 0; r < 8; ++r) {
                float v0 = c0[r] + bb0, v1 = c1[r] + bb1;
                if (mat < 2) {                          // cosine-attn normalize
                    float ss = v0 * v0 + v1 * v1;
                    ss += __shfl_xor(ss, 1, 32);
                    ss += __shfl_xor(ss, 2, 32);
                    ss += __shfl_xor(ss, 4, 32);
                    ss += __shfl_xor(ss, 8, 32);
                    const float rn = rsqrtf(fmaxf(ss, 1e-24f)) * qs;
                    v0 *= rn; v1 *= rn;
                }
                const int row = mt * 16 + half * 8 + r;
                dst[row * HD + col]      = (bf16_t)v0;
                dst[row * HD + 16 + col] = (bf16_t)v1;
            }
        }
    }
}

// ---------------------------------------------------------------------------
// Kernel 2: attention core — one wave per (window, head)
// ---------------------------------------------------------------------------
__global__ __launch_bounds__(256) void k_attn(
    const bf16_t* __restrict__ qkvn, const float* __restrict__ biasw,
    bf16_t* __restrict__ attn_out) {
    __shared__ bf16_t sc[HEADS * NPAD * NPAD];  // 64 KB: per-head prob matrix
    const int win  = blockIdx.x;
    const int head = threadIdx.x >> 5;
    const int lane = threadIdx.x & 31;
    const int col  = lane & 15;
    const int half = lane >> 4;

    const bf16_t* qn = qkvn + (((size_t)win * 3 + 0) * HEADS + head) * (NPAD * HD);
    const bf16_t* kn = qn + (size_t)HEADS * NPAD * HD;
    const bf16_t* vv = qn + (size_t)2 * HEADS * NPAD * HD;

    v16bf aq[4], bk[4];
#pragma unroll
    for (int t = 0; t < 4; ++t) {
        aq[t] = load_a_frag(qn + t * 16 * HD, HD);
        bk[t] = load_b_frag_nk(kn + t * 16 * HD, HD);   // scores = qn . kn
    }
    v8f s[4][4];
#pragma unroll
    for (int tr = 0; tr < 4; ++tr)
#pragma unroll
        for (int tc = 0; tc < 4; ++tc) {
            v8f z = {};
            s[tr][tc] = wmma_bf16(aq[tr], bk[tc], z);
        }

    bf16_t* myp = sc + head * (NPAD * NPAD);
#pragma unroll
    for (int tr = 0; tr < 4; ++tr) {
#pragma unroll
        for (int r = 0; r < 8; ++r) {
            const int row = tr * 16 + half * 8 + r;
            float l[4];
            float m = -1e30f;
#pragma unroll
            for (int tc = 0; tc < 4; ++tc) {
                const int cg = tc * 16 + col;
                float lg = s[tr][tc][r];
                if (cg < NTOK) {
                    if (row < NTOK) lg += biasw[head * (NTOK * NTOK) + row * NTOK + cg];
                } else {
                    lg = -1e30f;                        // mask padded cols
                }
                l[tc] = lg;
                m = fmaxf(m, lg);
            }
            m = fmaxf(m, __shfl_xor(m, 1, 32));
            m = fmaxf(m, __shfl_xor(m, 2, 32));
            m = fmaxf(m, __shfl_xor(m, 4, 32));
            m = fmaxf(m, __shfl_xor(m, 8, 32));
            float sum = 0.f;
#pragma unroll
            for (int tc = 0; tc < 4; ++tc) { l[tc] = __expf(l[tc] - m); sum += l[tc]; }
            sum += __shfl_xor(sum, 1, 32);
            sum += __shfl_xor(sum, 2, 32);
            sum += __shfl_xor(sum, 4, 32);
            sum += __shfl_xor(sum, 8, 32);
            const float inv = 1.f / sum;
#pragma unroll
            for (int tc = 0; tc < 4; ++tc)
                myp[row * NPAD + tc * 16 + col] = (bf16_t)(l[tc] * inv);
        }
    }

    // ctx = P @ V  (A from LDS, V as [K][N] B-fragments)
    v16bf bv[2][2];
#pragma unroll
    for (int kt = 0; kt < 2; ++kt)
#pragma unroll
        for (int nt = 0; nt < 2; ++nt)
            bv[kt][nt] = load_b_frag_kn(vv + kt * 32 * HD + nt * 16, HD);

    bf16_t* outw = attn_out + (size_t)win * NPAD * CH;
#pragma unroll
    for (int tr = 0; tr < 4; ++tr)
#pragma unroll
        for (int nt = 0; nt < 2; ++nt) {
            v8f acc = {};
#pragma unroll
            for (int kt = 0; kt < 2; ++kt) {
                const v16bf ap = load_a_frag(myp + tr * 16 * NPAD + kt * 32, NPAD);
                acc = wmma_bf16(ap, bv[kt][nt], acc);
            }
            store_c_bf16(outw + tr * 16 * CH + head * HD + nt * 16, CH, acc);
        }
}

// ---------------------------------------------------------------------------
// Kernel 3: output projection + residual, scatter back to NCHW (y0 = x + proj)
// ---------------------------------------------------------------------------
__global__ __launch_bounds__(256) void k_proj(
    const float* __restrict__ x, const bf16_t* __restrict__ attn,
    const bf16_t* __restrict__ projw, const float* __restrict__ proj_b,
    float* __restrict__ y0) {
    __shared__ bf16_t aw[NPAD * 264];
    const int win = blockIdx.x;
    stage_tile_async(aw, attn + (size_t)win * NPAD * CH, CH * 2u);
    __syncthreads();

    const int wave = threadIdx.x >> 5;
    const int lane = threadIdx.x & 31;
    const int col  = lane & 15;
    const int half = lane >> 4;
    const int bi = win >> 6, widx = win & 63;
    const int wy = widx >> 3, wx = widx & 7;

    for (int sub = 0; sub < 2; ++sub) {              // 2 N-tiles per wave
        const int nt = wave * 2 + sub;               // 16 tiles of 16 channels
        __builtin_prefetch(projw + (size_t)(nt * 16) * 256, 0, 1);
        for (int mt = 0; mt < 4; ++mt) {
            v8f acc = {};
#pragma unroll
            for (int k = 0; k < 8; ++k) {
                const v16bf a = load_a_frag(&aw[mt * 16 * 264 + k * 32], 264);
                const v16bf b = load_b_frag_nk(projw + (size_t)(nt * 16) * 256 + k * 32, 256);
                acc = wmma_bf16(a, b, acc);
            }
            const int ch = nt * 16 + col;
            const float pb = proj_b[ch];
#pragma unroll
            for (int r = 0; r < 8; ++r) {
                const int row = mt * 16 + half * 8 + r;
                if (row < NTOK) {
                    const int ty = row / WS, tx = row % WS;
                    const int sp = (wy * WS + ty) * RES + (wx * WS + tx);
                    const size_t o = ((size_t)bi * CH + ch) * SPAT + sp;
                    y0[o] = x[o] + acc[r] + pb;
                }
            }
        }
    }
}

// ---------------------------------------------------------------------------
// GroupNorm kernels: each (b,g) region is a contiguous 50176-float slab.
// ---------------------------------------------------------------------------
__device__ __forceinline__ void gn_stats(const float* __restrict__ p,
                                         float* red, float* red2,
                                         float& mean, float& rs) {
    float s = 0.f, ss = 0.f;
    for (int i = threadIdx.x; i < 16 * SPAT; i += 256) {
        const float v = p[i];
        s += v; ss += v * v;
    }
    red[threadIdx.x] = s; red2[threadIdx.x] = ss;
    __syncthreads();
    for (int st = 128; st > 0; st >>= 1) {
        if ((int)threadIdx.x < st) {
            red[threadIdx.x]  += red[threadIdx.x + st];
            red2[threadIdx.x] += red2[threadIdx.x + st];
        }
        __syncthreads();
    }
    mean = red[0] * (1.f / (16.f * SPAT));
    const float var = red2[0] * (1.f / (16.f * SPAT)) - mean * mean;
    rs = rsqrtf(var + 1e-5f);
}

__global__ __launch_bounds__(256) void k_gn_relu(
    const float* __restrict__ in, const float* __restrict__ w,
    const float* __restrict__ bi, float* __restrict__ out) {
    __shared__ float red[256], red2[256];
    const int bg = blockIdx.x;                       // b*16 + g
    const float* p = in + (size_t)bg * 16 * SPAT;
    float mean, rs;
    gn_stats(p, red, red2, mean, rs);
    const int g = bg & 15;
    float* q = out + (size_t)bg * 16 * SPAT;
    for (int i = threadIdx.x; i < 16 * SPAT; i += 256) {
        const int c = (g << 4) + (i / SPAT);
        q[i] = fmaxf((p[i] - mean) * rs * w[c] + bi[c], 0.f);
    }
}

__global__ __launch_bounds__(256) void k_gn_to_bf(
    const float* __restrict__ in, const float* __restrict__ w,
    const float* __restrict__ bi, bf16_t* __restrict__ mbf) {
    __shared__ float red[256], red2[256];
    const int bg = blockIdx.x;
    const float* p = in + (size_t)bg * 16 * SPAT;
    float mean, rs;
    gn_stats(p, red, red2, mean, rs);
    const int b = bg >> 4, g = bg & 15;
    for (int i = threadIdx.x; i < 16 * SPAT; i += 256) {
        const int cl = i / SPAT, sp = i % SPAT;
        const int c = (g << 4) + cl;
        const float v = (p[i] - mean) * rs * w[c] + bi[c];
        mbf[((size_t)b * SPAT + sp) * CH + c] = (bf16_t)v;   // [pixel][C] GEMM layout
    }
}

// ---------------------------------------------------------------------------
// MLP GEMM 1: [PIX,256] @ W1^T[256,1024] + b1, ReLU -> bf16
// ---------------------------------------------------------------------------
__global__ __launch_bounds__(256) void k_mlp1(
    const bf16_t* __restrict__ A, const bf16_t* __restrict__ W1,
    const float* __restrict__ b1, bf16_t* __restrict__ out) {
    __shared__ bf16_t as[64 * 264];
    const int mb = blockIdx.y, nb = blockIdx.x;
    stage_tile_async(as, A + (size_t)mb * 64 * CH, CH * 2u);
    __syncthreads();

    const int wave = threadIdx.x >> 5;
    const int lane = threadIdx.x & 31;
    const int col  = lane & 15;
    const int half = lane >> 4;
    const int nt = nb * 8 + wave;                    // of 64 col tiles
    __builtin_prefetch(W1 + (size_t)(nt * 16) * 256, 0, 1);
    v8f acc[4] = {{}, {}, {}, {}};
#pragma unroll
    for (int k = 0; k < 8; ++k) {
        const v16bf b = load_b_frag_nk(W1 + (size_t)(nt * 16) * 256 + k * 32, 256);
#pragma unroll
        for (int mt = 0; mt < 4; ++mt) {
            const v16bf a = load_a_frag(&as[mt * 16 * 264 + k * 32], 264);
            acc[mt] = wmma_bf16(a, b, acc[mt]);
        }
    }
    const float bb = b1[nt * 16 + col];
#pragma unroll
    for (int mt = 0; mt < 4; ++mt)
#pragma unroll
        for (int r = 0; r < 8; ++r) {
            const size_t row = (size_t)mb * 64 + mt * 16 + half * 8 + r;
            out[row * MLPD + nt * 16 + col] = (bf16_t)fmaxf(acc[mt][r] + bb, 0.f);
        }
}

// ---------------------------------------------------------------------------
// MLP GEMM 2: [PIX,1024] @ W2^T[1024,256] + b2 + h -> t (NCHW f32)
// ---------------------------------------------------------------------------
__global__ __launch_bounds__(256) void k_mlp2(
    const bf16_t* __restrict__ A, const bf16_t* __restrict__ W2,
    const float* __restrict__ b2, const float* __restrict__ h,
    float* __restrict__ t) {
    __shared__ bf16_t as[64 * 264];
    const int mb = blockIdx.y, nb = blockIdx.x;
    const int wave = threadIdx.x >> 5;
    const int lane = threadIdx.x & 31;
    const int col  = lane & 15;
    const int half = lane >> 4;
    const int nt = nb * 8 + wave;                    // of 16 col tiles
    __builtin_prefetch(W2 + (size_t)(nt * 16) * MLPD, 0, 1);
    v8f acc[4] = {{}, {}, {}, {}};

    for (int kc = 0; kc < 4; ++kc) {                 // K = 1024 in 256 chunks
        __syncthreads();
        stage_tile_async(as, A + (size_t)mb * 64 * MLPD + kc * 256, MLPD * 2u);
        __syncthreads();
#pragma unroll
        for (int k = 0; k < 8; ++k) {
            const v16bf b = load_b_frag_nk(W2 + (size_t)(nt * 16) * MLPD + kc * 256 + k * 32, MLPD);
#pragma unroll
            for (int mt = 0; mt < 4; ++mt) {
                const v16bf a = load_a_frag(&as[mt * 16 * 264 + k * 32], 264);
                acc[mt] = wmma_bf16(a, b, acc[mt]);
            }
        }
    }
    const int c = nt * 16 + col;
    const float bb = b2[c];
#pragma unroll
    for (int mt = 0; mt < 4; ++mt)
#pragma unroll
        for (int r = 0; r < 8; ++r) {
            const int row = mb * 64 + mt * 16 + half * 8 + r;
            const int bi = row / SPAT, sp = row % SPAT;
            const size_t o = ((size_t)bi * CH + c) * SPAT + sp;
            t[o] = h[o] + acc[mt][r] + bb;
        }
}

// ---------------------------------------------------------------------------
// Launch
// ---------------------------------------------------------------------------
extern "C" void kernel_launch(void* const* d_in, const int* in_sizes, int n_in,
                              void* d_out, int out_size, void* d_ws, size_t ws_size,
                              hipStream_t stream) {
    const float* x           = (const float*)d_in[0];
    const float* qkv_w       = (const float*)d_in[1];
    const float* qkv_b       = (const float*)d_in[2];
    const float* logit_scale = (const float*)d_in[3];
    const float* bias_table  = (const float*)d_in[4];
    const int*   rel_index   = (const int*)  d_in[5];
    const float* proj_w      = (const float*)d_in[6];
    const float* proj_b      = (const float*)d_in[7];
    const float* gn1_w       = (const float*)d_in[8];
    const float* gn1_b       = (const float*)d_in[9];
    const float* gn2_w       = (const float*)d_in[10];
    const float* gn2_b       = (const float*)d_in[11];
    const float* mlp_gn_w    = (const float*)d_in[12];
    const float* mlp_gn_b    = (const float*)d_in[13];
    const float* mlp_w1      = (const float*)d_in[14];
    const float* mlp_b1      = (const float*)d_in[15];
    const float* mlp_w2      = (const float*)d_in[16];
    const float* mlp_b2      = (const float*)d_in[17];
    float* out = (float*)d_out;

    char* ws = (char*)d_ws;
    bf16_t* qkvw   = (bf16_t*)(ws + OFF_QKVW);
    bf16_t* projw  = (bf16_t*)(ws + OFF_PROJW);
    bf16_t* w1b    = (bf16_t*)(ws + OFF_W1);
    bf16_t* w2b    = (bf16_t*)(ws + OFF_W2);
    float*  biasw  = (float*) (ws + OFF_BIAS);
    float*  qscale = (float*) (ws + OFF_QSCALE);
    bf16_t* qkvn   = (bf16_t*)(ws + OFF_QKVN);
    bf16_t* attn   = (bf16_t*)(ws + OFF_ATTN);
    bf16_t* mbf    = (bf16_t*)(ws + OFF_ATTN);   // reuse (attn consumed by k_proj)
    float*  y0     = (float*) (ws + OFF_Y0);
    float*  tbuf   = (float*) (ws + OFF_Y0);     // reuse (y0 consumed by GN1)
    float*  hbuf   = (float*) (ws + OFF_H);
    bf16_t* m1     = (bf16_t*)(ws + OFF_M1);

    k_prep<<<96, 256, 0, stream>>>(qkv_w, proj_w, mlp_w1, mlp_w2, bias_table,
                                   rel_index, logit_scale, qkvw, projw, w1b, w2b,
                                   biasw, qscale);
    k_qkv <<<NWIN, 256, 0, stream>>>(x, qkv_b, qkvw, qscale, qkvn);
    k_attn<<<NWIN, 256, 0, stream>>>(qkvn, biasw, attn);
    k_proj<<<NWIN, 256, 0, stream>>>(x, attn, projw, proj_b, y0);
    k_gn_relu<<<BATCH * GROUPS, 256, 0, stream>>>(y0, gn1_w, gn1_b, hbuf);
    k_gn_to_bf<<<BATCH * GROUPS, 256, 0, stream>>>(hbuf, mlp_gn_w, mlp_gn_b, mbf);
    k_mlp1<<<dim3(8, PIX / 64), 256, 0, stream>>>(mbf, w1b, mlp_b1, m1);
    k_mlp2<<<dim3(2, PIX / 64), 256, 0, stream>>>(m1, w2b, mlp_b2, hbuf, tbuf);
    k_gn_relu<<<BATCH * GROUPS, 256, 0, stream>>>(tbuf, gn2_w, gn2_b, out);
    (void)in_sizes; (void)n_in; (void)out_size; (void)ws_size;
}
